// ClassifierModel_18794776887872
// MI455X (gfx1250) — matrix operand order repaired and checked by reading, not compiled, not run on hardware
//
#include <hip/hip_runtime.h>
#include <hip/hip_bf16.h>
#include <math.h>

// ---------------------------------------------------------------------------
// Model constants
// ---------------------------------------------------------------------------
constexpr int HID = 768, NH = 12, HD = 64, NL = 12, FFND = 3072;
constexpr int WSZ = 256, SEQ = 4096, NBATCH = 2, NCL = 43;
constexpr int TOK = NBATCH * SEQ;          // 8192 tokens
constexpr int NBLK = SEQ / WSZ;            // 16 blocks
constexpr float NEGV = -1.0e9f;

// ---------------------------------------------------------------------------
// CDNA5 WMMA types / helpers (wave32, v_wmma_f32_16x16x32_bf16)
// ---------------------------------------------------------------------------
typedef __bf16 v16bf __attribute__((ext_vector_type(16)));
typedef __bf16 bfx8  __attribute__((ext_vector_type(8)));
typedef float  v8f   __attribute__((ext_vector_type(8)));

__device__ inline v8f wmma_bf16(v16bf a, v16bf b, v8f c) {
  return __builtin_amdgcn_wmma_f32_16x16x32_bf16(false, a, false, b,
                                                 (short)0, c, false, false);
}

// two 16-byte vector loads -> one 16-element bf16 fragment
__device__ inline v16bf ld2x8(const __bf16* p0, const __bf16* p1) {
  bfx8 lo = *(const bfx8*)p0;
  bfx8 hi = *(const bfx8*)p1;
  return __builtin_shufflevector(lo, hi, 0, 1, 2, 3, 4, 5, 6, 7,
                                 8, 9, 10, 11, 12, 13, 14, 15);
}

__device__ inline float gelu_exact(float v) {
  return 0.5f * v * (1.0f + erff(v * 0.70710678118654752f));
}

// ---------------------------------------------------------------------------
// Weight transpose + f32->bf16 convert:  W[K][N] f32  ->  Wt[N][K] bf16
// ---------------------------------------------------------------------------
__global__ __launch_bounds__(256)
void wt_convert_kernel(const float* __restrict__ W, __bf16* __restrict__ Wt,
                       int K, int N) {
  __shared__ float tile[32][33];
  const int k0 = blockIdx.x * 32;
  const int n0 = blockIdx.y * 32;
  const int tx = threadIdx.x & 31;
  const int ty = threadIdx.x >> 5;
#pragma unroll
  for (int r = ty; r < 32; r += 8)
    tile[r][tx] = W[(size_t)(k0 + r) * N + (n0 + tx)];
  __syncthreads();
#pragma unroll
  for (int r = ty; r < 32; r += 8)
    Wt[(size_t)(n0 + r) * K + (k0 + tx)] = (__bf16)tile[tx][r];
}

// ---------------------------------------------------------------------------
// Embedding gather + LayerNorm  (one block per token)
// ---------------------------------------------------------------------------
__global__ __launch_bounds__(256)
void embed_ln_kernel(const int* __restrict__ ids,
                     const float* __restrict__ wemb,
                     const float* __restrict__ pemb,
                     const float* __restrict__ g, const float* __restrict__ bta,
                     float* __restrict__ xf, __bf16* __restrict__ xb) {
  const int tok = blockIdx.x;
  const int s = tok % SEQ;
  __shared__ float red[256];
  __shared__ float buf[HID];
  const int id = ids[tok];
  const float* wp = wemb + (size_t)id * HID;
  const float* pp = pemb + (size_t)s * HID;
  float part = 0.f;
  for (int j = threadIdx.x; j < HID; j += 256) {
    float v = wp[j] + pp[j];
    buf[j] = v; part += v;
  }
  red[threadIdx.x] = part; __syncthreads();
  for (int st = 128; st > 0; st >>= 1) {
    if (threadIdx.x < st) red[threadIdx.x] += red[threadIdx.x + st];
    __syncthreads();
  }
  const float mean = red[0] / (float)HID; __syncthreads();
  part = 0.f;
  for (int j = threadIdx.x; j < HID; j += 256) {
    float d = buf[j] - mean; part += d * d;
  }
  red[threadIdx.x] = part; __syncthreads();
  for (int st = 128; st > 0; st >>= 1) {
    if (threadIdx.x < st) red[threadIdx.x] += red[threadIdx.x + st];
    __syncthreads();
  }
  const float rstd = rsqrtf(red[0] / (float)HID + 1e-5f);
  const size_t base = (size_t)tok * HID;
  for (int j = threadIdx.x; j < HID; j += 256) {
    float v = (buf[j] - mean) * rstd * g[j] + bta[j];
    xf[base + j] = v;
    xb[base + j] = (__bf16)v;
  }
}

// ---------------------------------------------------------------------------
// Residual add + LayerNorm   out = LN(xres + delta)
// ---------------------------------------------------------------------------
__global__ __launch_bounds__(256)
void add_ln_kernel(const float* __restrict__ xres, const float* __restrict__ delta,
                   const float* __restrict__ g, const float* __restrict__ bta,
                   float* __restrict__ outf, __bf16* __restrict__ outb) {
  const int tok = blockIdx.x;
  __shared__ float red[256];
  __shared__ float buf[HID];
  const size_t base = (size_t)tok * HID;
  float part = 0.f;
  for (int j = threadIdx.x; j < HID; j += 256) {
    float v = xres[base + j] + delta[base + j];
    buf[j] = v; part += v;
  }
  red[threadIdx.x] = part; __syncthreads();
  for (int st = 128; st > 0; st >>= 1) {
    if (threadIdx.x < st) red[threadIdx.x] += red[threadIdx.x + st];
    __syncthreads();
  }
  const float mean = red[0] / (float)HID; __syncthreads();
  part = 0.f;
  for (int j = threadIdx.x; j < HID; j += 256) {
    float d = buf[j] - mean; part += d * d;
  }
  red[threadIdx.x] = part; __syncthreads();
  for (int st = 128; st > 0; st >>= 1) {
    if (threadIdx.x < st) red[threadIdx.x] += red[threadIdx.x + st];
    __syncthreads();
  }
  const float rstd = rsqrtf(red[0] / (float)HID + 1e-5f);
  for (int j = threadIdx.x; j < HID; j += 256) {
    float v = (buf[j] - mean) * rstd * g[j] + bta[j];
    outf[base + j] = v;
    outb[base + j] = (__bf16)v;
  }
}

// ---------------------------------------------------------------------------
// WMMA GEMM:  C[M,N] = act( alpha * (A_bf16[M,K] @ W[K,N] + bias) )
//   Wt[N][K] bf16 pre-transposed; per K-step: batch 12 b128 loads, then
//   8 back-to-back WMMAs (staged waits, latency overlap).
//   8 waves/block: 4 on M (32 rows) x 2 on N (64 cols) -> 128x128 tile
//   storeMode 0: Cb/Cf row-major;  storeMode 1: Cb -> vT[(b*HID+col)*SEQ + s]
// ---------------------------------------------------------------------------
__global__ __launch_bounds__(256)
void gemm_bf16_kernel(const __bf16* __restrict__ A, long lda,
                      const __bf16* __restrict__ Wt, int K, int N,
                      const float* __restrict__ bias,
                      float alpha, int act, int M,
                      float* __restrict__ Cf, __bf16* __restrict__ Cb, long ldc,
                      int storeMode) {
  const int lane  = threadIdx.x & 31;
  const int wave  = threadIdx.x >> 5;
  const int wm    = wave & 3;
  const int wn    = wave >> 2;
  const int mbase = blockIdx.x * 128 + wm * 32;
  const int nbase = blockIdx.y * 128 + wn * 64;
  const int l16   = lane & 15;
  const int lhalf = lane >> 4;

  v8f acc[2][4];
#pragma unroll
  for (int u = 0; u < 2; ++u)
#pragma unroll
    for (int t = 0; t < 4; ++t)
#pragma unroll
      for (int i = 0; i < 8; ++i) acc[u][t][i] = 0.f;

  int arow0 = mbase + l16;      if (arow0 >= M) arow0 = M - 1;
  int arow1 = mbase + 16 + l16; if (arow1 >= M) arow1 = M - 1;
  const __bf16* ap0 = A + (size_t)arow0 * lda;
  const __bf16* ap1 = A + (size_t)arow1 * lda;
  // per-lane base pointers into the 4 B columns this lane feeds
  const __bf16* bp0 = Wt + (size_t)(nbase + 0 * 16 + l16) * K + lhalf * 16;
  const __bf16* bp1 = Wt + (size_t)(nbase + 1 * 16 + l16) * K + lhalf * 16;
  const __bf16* bp2 = Wt + (size_t)(nbase + 2 * 16 + l16) * K + lhalf * 16;
  const __bf16* bp3 = Wt + (size_t)(nbase + 3 * 16 + l16) * K + lhalf * 16;

  for (int k0 = 0; k0 < K; k0 += 32) {
    const int kb8 = k0 + lhalf * 16;
    // ---- batch ALL loads for this K-step (A: 4 x b128, B: 8 x b128) ----
    const v16bf a0 = ld2x8(ap0 + kb8, ap0 + kb8 + 8);
    const v16bf a1 = ld2x8(ap1 + kb8, ap1 + kb8 + 8);
    v16bf bfrag[4];
    bfrag[0] = ld2x8(bp0 + k0, bp0 + k0 + 8);
    bfrag[1] = ld2x8(bp1 + k0, bp1 + k0 + 8);
    bfrag[2] = ld2x8(bp2 + k0, bp2 + k0 + 8);
    bfrag[3] = ld2x8(bp3 + k0, bp3 + k0 + 8);
    if (k0 + 32 < K) __builtin_prefetch(bp0 + k0 + 32, 0, 3);
    // ---- then 8 back-to-back WMMAs ----
#pragma unroll
    for (int t = 0; t < 4; ++t) {
      acc[0][t] = wmma_bf16(a0, bfrag[t], acc[0][t]);
      acc[1][t] = wmma_bf16(a1, bfrag[t], acc[1][t]);
    }
  }

  // epilogue: element r of acc[u][t] is (row = mbase + u*16 + 8*lhalf + r, col)
#pragma unroll
  for (int u = 0; u < 2; ++u)
#pragma unroll
    for (int t = 0; t < 4; ++t) {
      const int col = nbase + t * 16 + l16;
      const float bcol = bias[col];
#pragma unroll
      for (int r = 0; r < 8; ++r) {
        const int row = mbase + u * 16 + 8 * lhalf + r;
        if (row < M) {
          float v = alpha * (acc[u][t][r] + bcol);
          if (act == 1) v = gelu_exact(v);
          if (storeMode == 0) {
            if (Cf) Cf[(size_t)row * ldc + col] = v;
            if (Cb) Cb[(size_t)row * ldc + col] = (__bf16)v;
          } else {  // transposed attention layout (b, h, d, s)
            const int b = row / SEQ, s = row % SEQ;
            Cb[((size_t)b * HID + col) * SEQ + s] = (__bf16)v;
          }
        }
      }
    }
}

// ---------------------------------------------------------------------------
// Banded flash attention (+ global column vs key 0), one wave per 16-q tile.
//   q,k bf16 token-major [(b*SEQ+s)*HID + h*HD + d]
//   v  bf16 transposed   [(b*HID + h*HD + d)*SEQ + s]
//   grid = (WSZ/16, NBLK, NBATCH*NH), block = 32
// ---------------------------------------------------------------------------
__device__ inline bool band_valid(int qi, int ki, int nb, int b,
                                  const int* __restrict__ mask) {
  const int kpos = nb * WSZ - WSZ + ki;
  const bool rel_ok = (ki >= qi) && (ki <= qi + 2 * WSZ);
  const bool pos_ok = (kpos >= 0) && (kpos < SEQ) && (kpos != 0);
  const int kc = kpos < 0 ? 0 : (kpos >= SEQ ? SEQ - 1 : kpos);
  const bool pad_ok = mask[b * SEQ + kc] > 0;
  return rel_ok && pos_ok && pad_ok;
}

__global__ __launch_bounds__(32)
void flash_band_kernel(const __bf16* __restrict__ qb,
                       const __bf16* __restrict__ kbf,
                       const __bf16* __restrict__ vT,
                       const int* __restrict__ mask,
                       float* __restrict__ attnf) {
  const int lane  = threadIdx.x & 31;
  const int l16   = lane & 15;
  const int lhalf = lane >> 4;
  const int qt = blockIdx.x;
  const int nb = blockIdx.y;
  const int bh = blockIdx.z;
  const int b = bh / NH, h = bh % NH;

  __shared__ __bf16 probs[16][40];

  // Q fragments (two 32-wide d halves)
  v16bf aq[2];
  {
    const int srow = nb * WSZ + qt * 16 + l16;
    const __bf16* qp = qb + (size_t)(b * SEQ + srow) * HID + h * HD;
#pragma unroll
    for (int t = 0; t < 2; ++t) {
      const int kb8 = t * 32 + lhalf * 16;
      aq[t] = ld2x8(qp + kb8, qp + kb8 + 8);
    }
  }

  float mrow[8], lsum[8];
  v8f acc[4];
#pragma unroll
  for (int r = 0; r < 8; ++r) { mrow[r] = -3.0e38f; lsum[r] = 0.f; }
#pragma unroll
  for (int t = 0; t < 4; ++t)
#pragma unroll
    for (int i = 0; i < 8; ++i) acc[t][i] = 0.f;

  const int qi0 = qt * 16;

  for (int kb = 0; kb < 3 * WSZ; kb += 32) {
    // --- batch the 4 K fragments (2 key groups x 2 d-halves), then 4 WMMAs
    v16bf bk[4];
    {
      const int ki0 = kb + l16;
      const int kpos0 = nb * WSZ - WSZ + ki0;
      const int kc0 = kpos0 < 0 ? 0 : (kpos0 >= SEQ ? SEQ - 1 : kpos0);
      const __bf16* kp0 = kbf + (size_t)(b * SEQ + kc0) * HID + h * HD;
      const int ki1 = kb + 16 + l16;
      const int kpos1 = nb * WSZ - WSZ + ki1;
      const int kc1 = kpos1 < 0 ? 0 : (kpos1 >= SEQ ? SEQ - 1 : kpos1);
      const __bf16* kp1 = kbf + (size_t)(b * SEQ + kc1) * HID + h * HD;
#pragma unroll
      for (int t = 0; t < 2; ++t) {
        const int off = t * 32 + lhalf * 16;
        bk[t]     = ld2x8(kp0 + off, kp0 + off + 8);
        bk[2 + t] = ld2x8(kp1 + off, kp1 + off + 8);
      }
    }
    v8f s0, s1;
#pragma unroll
    for (int i = 0; i < 8; ++i) { s0[i] = 0.f; s1[i] = 0.f; }
    s0 = wmma_bf16(aq[0], bk[0], s0);
    s0 = wmma_bf16(aq[1], bk[1], s0);
    s1 = wmma_bf16(aq[0], bk[2], s1);
    s1 = wmma_bf16(aq[1], bk[3], s1);

    // --- mask + online softmax update
    float p0[8], p1[8];
#pragma unroll
    for (int r = 0; r < 8; ++r) {
      const int qi = qi0 + 8 * lhalf + r;
      const int ki0 = kb + l16, ki1 = kb + 16 + l16;
      float v0 = band_valid(qi, ki0, nb, b, mask) ? s0[r] : NEGV;
      float v1 = band_valid(qi, ki1, nb, b, mask) ? s1[r] : NEGV;
      float mx = fmaxf(v0, v1);
#pragma unroll
      for (int off = 1; off < 16; off <<= 1) mx = fmaxf(mx, __shfl_xor(mx, off, 32));
      const float nm = fmaxf(mrow[r], mx);
      const float sc = __expf(mrow[r] - nm);
      const float e0 = __expf(v0 - nm), e1 = __expf(v1 - nm);
      float ps = e0 + e1;
#pragma unroll
      for (int off = 1; off < 16; off <<= 1) ps += __shfl_xor(ps, off, 32);
      lsum[r] = lsum[r] * sc + ps;
      mrow[r] = nm;
#pragma unroll
      for (int t = 0; t < 4; ++t) acc[t][r] *= sc;
      p0[r] = e0; p1[r] = e1;
    }

    // --- stage probs through LDS (C layout -> A layout)
#pragma unroll
    for (int r = 0; r < 8; ++r) {
      const int rowi = 8 * lhalf + r;
      probs[rowi][l16]      = (__bf16)p0[r];
      probs[rowi][16 + l16] = (__bf16)p1[r];
    }
    __syncthreads();
    v16bf apf;
#pragma unroll
    for (int e = 0; e < 8; ++e) apf[e]     = probs[l16][lhalf * 16 + e];
#pragma unroll
    for (int e = 0; e < 8; ++e) apf[8 + e] = probs[l16][lhalf * 16 + 8 + e];
    __syncthreads();

    // --- PV with transposed V: batch the 4 V fragments, then 4 WMMAs
    {
      int sbase = nb * WSZ - WSZ + kb + lhalf * 16;
      if (sbase < 0) sbase = 0;   // whole group invalid -> probs are 0
      v16bf bv[4];
#pragma unroll
      for (int t2 = 0; t2 < 4; ++t2) {
        const int d = t2 * 16 + l16;
        const __bf16* vp = vT + (size_t)(b * HID + h * HD + d) * SEQ + sbase;
        bv[t2] = ld2x8(vp, vp + 8);
      }
#pragma unroll
      for (int t2 = 0; t2 < 4; ++t2)
        acc[t2] = wmma_bf16(apf, bv[t2], acc[t2]);
    }
  }

  // --- global column: every row also attends to key position 0
  float sext;
  {
    const int row = l16;
    const int srow = nb * WSZ + qt * 16 + row;
    const __bf16* qp = qb + (size_t)(b * SEQ + srow) * HID + h * HD;
    const __bf16* kp = kbf + (size_t)(b * SEQ) * HID + h * HD;
    float sum = 0.f;
#pragma unroll 8
    for (int d = 0; d < HD; ++d) sum += (float)qp[d] * (float)kp[d];
    sext = sum;
  }
  float v0d[4];
#pragma unroll
  for (int t = 0; t < 4; ++t)
    v0d[t] = (float)vT[(size_t)(b * HID + h * HD + t * 16 + l16) * SEQ];
#pragma unroll
  for (int r = 0; r < 8; ++r) {
    const int rowi = 8 * lhalf + r;
    const float se = __shfl(sext, rowi, 32);
    const float nm = fmaxf(mrow[r], se);
    const float sc = __expf(mrow[r] - nm);
    const float pe = __expf(se - nm);
    lsum[r] = lsum[r] * sc + pe;
    mrow[r] = nm;
#pragma unroll
    for (int t = 0; t < 4; ++t) acc[t][r] = acc[t][r] * sc + pe * v0d[t];
  }

  // --- normalized write (token-major HID layout)
#pragma unroll
  for (int t = 0; t < 4; ++t)
#pragma unroll
    for (int r = 0; r < 8; ++r) {
      const int srow = nb * WSZ + qt * 16 + 8 * lhalf + r;
      const int d = t * 16 + l16;
      attnf[(size_t)(b * SEQ + srow) * HID + h * HD + d] = acc[t][r] / lsum[r];
    }
}

// ---------------------------------------------------------------------------
// Global attention for token 0:  og[b,h,:] = softmax(qg . kg^T) @ vg
// ---------------------------------------------------------------------------
__global__ __launch_bounds__(256)
void global_attn_kernel(const float* __restrict__ qg, const float* __restrict__ kg,
                        const float* __restrict__ vg, const int* __restrict__ mask,
                        float* __restrict__ og) {
  const int bh = blockIdx.x;
  const int b = bh / NH, h = bh % NH;
  __shared__ float sc[SEQ];
  __shared__ float red[256];
  __shared__ float qs[HD];
  const int tid = threadIdx.x;
  if (tid < HD) qs[tid] = qg[(size_t)b * HID + h * HD + tid];
  __syncthreads();

  float lmax = -3.0e38f;
  for (int s = tid; s < SEQ; s += 256) {
    const float* kp = kg + (size_t)(b * SEQ + s) * HID + h * HD;
    float sum = 0.f;
#pragma unroll 8
    for (int d = 0; d < HD; ++d) sum += qs[d] * kp[d];
    const float v = (mask[b * SEQ + s] > 0) ? sum : NEGV;
    sc[s] = v;
    lmax = fmaxf(lmax, v);
  }
  red[tid] = lmax; __syncthreads();
  for (int st = 128; st > 0; st >>= 1) {
    if (tid < st) red[tid] = fmaxf(red[tid], red[tid + st]);
    __syncthreads();
  }
  const float mx = red[0]; __syncthreads();

  float lsum = 0.f;
  for (int s = tid; s < SEQ; s += 256) {
    const float p = __expf(sc[s] - mx);
    sc[s] = p;
    lsum += p;
  }
  red[tid] = lsum; __syncthreads();
  for (int st = 128; st > 0; st >>= 1) {
    if (tid < st) red[tid] += red[tid + st];
    __syncthreads();
  }
  const float denom = red[0]; __syncthreads();

  const int d = tid & 63, qd = tid >> 6;
  float o = 0.f;
  for (int s = qd * (SEQ / 4); s < (qd + 1) * (SEQ / 4); ++s)
    o += sc[s] * vg[(size_t)(b * SEQ + s) * HID + h * HD + d];
  red[tid] = o; __syncthreads();
  if (tid < HD) {
    const float sum = red[tid] + red[tid + 64] + red[tid + 128] + red[tid + 192];
    og[(size_t)bh * HD + tid] = sum / denom;
  }
}

// ---------------------------------------------------------------------------
// Overwrite token-0 rows with og; emit bf16 mirror for the wo GEMM
// ---------------------------------------------------------------------------
__global__ __launch_bounds__(256)
void finalize_attn_kernel(float* __restrict__ attnf, __bf16* __restrict__ attnb,
                          const float* __restrict__ og) {
  const size_t i = (size_t)blockIdx.x * 256 + threadIdx.x;
  if (i >= (size_t)TOK * HID) return;
  const int tok = (int)(i / HID), c = (int)(i % HID);
  const int b = tok / SEQ, s = tok % SEQ;
  float v = attnf[i];
  if (s == 0) {
    const int h = c / HD, d = c % HD;
    v = og[(size_t)(b * NH + h) * HD + d];
    attnf[i] = v;
  }
  attnb[i] = (__bf16)v;
}

// ---------------------------------------------------------------------------
// Classifier head (tiny, one block)
// ---------------------------------------------------------------------------
__global__ __launch_bounds__(256)
void head_kernel(const float* __restrict__ xf,
                 const float* __restrict__ w1, const float* __restrict__ b1,
                 const float* __restrict__ w2, const float* __restrict__ b2,
                 float* __restrict__ out) {
  __shared__ float hbuf[NBATCH][512];
  for (int jj = threadIdx.x; jj < NBATCH * 512; jj += 256) {
    const int b = jj >> 9, c = jj & 511;
    const float* xp = xf + (size_t)(b * SEQ) * HID;
    float s = b1[c];
    for (int k = 0; k < HID; ++k) s += xp[k] * w1[(size_t)k * 512 + c];
    hbuf[b][c] = fmaxf(s, 0.f);
  }
  __syncthreads();
  for (int jj = threadIdx.x; jj < NBATCH * NCL; jj += 256) {
    const int b = jj / NCL, c = jj % NCL;
    float s = b2[c];
    for (int k = 0; k < 512; ++k) s += hbuf[b][k] * w2[(size_t)k * NCL + c];
    out[jj] = s;
  }
}

// ---------------------------------------------------------------------------
// Host-side orchestration
// ---------------------------------------------------------------------------
extern "C" void kernel_launch(void* const* d_in, const int* in_sizes, int n_in,
                              void* d_out, int out_size, void* d_ws, size_t ws_size,
                              hipStream_t stream) {
  (void)in_sizes; (void)n_in; (void)out_size; (void)ws_size;

  const int*   ids   = (const int*)d_in[0];
  const int*   amask = (const int*)d_in[1];
  const float* wemb  = (const float*)d_in[2];
  const float* pemb  = (const float*)d_in[3];
  const float* elng  = (const float*)d_in[4];
  const float* elnb  = (const float*)d_in[5];
  const float* Lwq  = (const float*)d_in[6];
  const float* Lbq  = (const float*)d_in[7];
  const float* Lwk  = (const float*)d_in[8];
  const float* Lbk  = (const float*)d_in[9];
  const float* Lwv  = (const float*)d_in[10];
  const float* Lbv  = (const float*)d_in[11];
  const float* Lwqg = (const float*)d_in[12];
  const float* Lbqg = (const float*)d_in[13];
  const float* Lwkg = (const float*)d_in[14];
  const float* Lbkg = (const float*)d_in[15];
  const float* Lwvg = (const float*)d_in[16];
  const float* Lbvg = (const float*)d_in[17];
  const float* Lwo  = (const float*)d_in[18];
  const float* Lbo  = (const float*)d_in[19];
  const float* Lln1g = (const float*)d_in[20];
  const float* Lln1b = (const float*)d_in[21];
  const float* Lw1  = (const float*)d_in[22];
  const float* Lb1  = (const float*)d_in[23];
  const float* Lw2  = (const float*)d_in[24];
  const float* Lb2  = (const float*)d_in[25];
  const float* Lln2g = (const float*)d_in[26];
  const float* Lln2b = (const float*)d_in[27];
  const float* hw1 = (const float*)d_in[28];
  const float* hb1 = (const float*)d_in[29];
  const float* hw2 = (const float*)d_in[30];
  const float* hb2 = (const float*)d_in[31];

  // workspace carve-up
  char* wp = (char*)d_ws;
  auto carve = [&](size_t bytes) -> void* {
    void* p = (void*)wp;
    wp += (bytes + 255) & ~(size_t)255;
    return p;
  };
  const size_t f32TH  = (size_t)TOK * HID * sizeof(float);
  const size_t bf16TH = (size_t)TOK * HID * sizeof(__bf16);
  const size_t wHH    = (size_t)HID * HID * sizeof(__bf16);
  const size_t wHF    = (size_t)HID * FFND * sizeof(__bf16);

  float*  xf    = (float*)carve(f32TH);
  __bf16* xb    = (__bf16*)carve(bf16TH);
  __bf16* qb    = (__bf16*)carve(bf16TH);
  __bf16* kb_   = (__bf16*)carve(bf16TH);
  __bf16* vTb   = (__bf16*)carve(bf16TH);      // V transposed (b,h,d,s)
  float*  kgf   = (float*)carve(f32TH);        // kg, reused for wo output
  float*  vgf   = (float*)carve(f32TH);        // vg, reused for FFN2 output
  float*  attnf = (float*)carve(f32TH);
  __bf16* attnb = (__bf16*)carve(bf16TH);
  float*  hf    = (float*)carve(f32TH);
  __bf16* hb    = (__bf16*)carve(bf16TH);
  __bf16* ffnb  = (__bf16*)carve((size_t)TOK * FFND * sizeof(__bf16));
  float*  qgf   = (float*)carve((size_t)NBATCH * HID * sizeof(float));
  float*  ogf   = (float*)carve((size_t)NBATCH * NH * HD * sizeof(float));
  // transposed bf16 weights for the current layer
  __bf16* tw_q  = (__bf16*)carve(wHH);
  __bf16* tw_k  = (__bf16*)carve(wHH);
  __bf16* tw_v  = (__bf16*)carve(wHH);
  __bf16* tw_qg = (__bf16*)carve(wHH);
  __bf16* tw_kg = (__bf16*)carve(wHH);
  __bf16* tw_vg = (__bf16*)carve(wHH);
  __bf16* tw_wo = (__bf16*)carve(wHH);
  __bf16* tw_w1 = (__bf16*)carve(wHF);
  __bf16* tw_w2 = (__bf16*)carve(wHF);

  const dim3 blk256(256);
  const dim3 gemmGridH(TOK / 128, HID / 128);      // 8192x768
  const dim3 gemmGridF(TOK / 128, FFND / 128);     // 8192x3072
  const dim3 gemmGridQG(1, HID / 128);             // 2x768
  const dim3 cvtHH(HID / 32, HID / 32);            // 768x768 transpose
  const dim3 cvtHF(HID / 32, FFND / 32);           // 768x3072
  const dim3 cvtFH(FFND / 32, HID / 32);           // 3072x768
  const float scale = 0.125f;                      // 1/sqrt(64)

  embed_ln_kernel<<<TOK, blk256, 0, stream>>>(ids, wemb, pemb, elng, elnb, xf, xb);

  for (int l = 0; l < NL; ++l) {
    const float* wq  = Lwq  + (size_t)l * HID * HID;
    const float* bq  = Lbq  + (size_t)l * HID;
    const float* wk  = Lwk  + (size_t)l * HID * HID;
    const float* bk  = Lbk  + (size_t)l * HID;
    const float* wv  = Lwv  + (size_t)l * HID * HID;
    const float* bv  = Lbv  + (size_t)l * HID;
    const float* wqg = Lwqg + (size_t)l * HID * HID;
    const float* bqg = Lbqg + (size_t)l * HID;
    const float* wkg = Lwkg + (size_t)l * HID * HID;
    const float* bkg = Lbkg + (size_t)l * HID;
    const float* wvg = Lwvg + (size_t)l * HID * HID;
    const float* bvg = Lbvg + (size_t)l * HID;
    const float* wo  = Lwo  + (size_t)l * HID * HID;
    const float* bo  = Lbo  + (size_t)l * HID;
    const float* l1g = Lln1g + (size_t)l * HID;
    const float* l1b = Lln1b + (size_t)l * HID;
    const float* w1  = Lw1  + (size_t)l * HID * FFND;
    const float* b1  = Lb1  + (size_t)l * FFND;
    const float* w2  = Lw2  + (size_t)l * FFND * HID;
    const float* b2  = Lb2  + (size_t)l * HID;
    const float* l2g = Lln2g + (size_t)l * HID;
    const float* l2b = Lln2b + (size_t)l * HID;

    // pre-transpose + convert this layer's GEMM weights to bf16 [N][K]
    wt_convert_kernel<<<cvtHH, blk256, 0, stream>>>(wq,  tw_q,  HID, HID);
    wt_convert_kernel<<<cvtHH, blk256, 0, stream>>>(wk,  tw_k,  HID, HID);
    wt_convert_kernel<<<cvtHH, blk256, 0, stream>>>(wv,  tw_v,  HID, HID);
    wt_convert_kernel<<<cvtHH, blk256, 0, stream>>>(wqg, tw_qg, HID, HID);
    wt_convert_kernel<<<cvtHH, blk256, 0, stream>>>(wkg, tw_kg, HID, HID);
    wt_convert_kernel<<<cvtHH, blk256, 0, stream>>>(wvg, tw_vg, HID, HID);
    wt_convert_kernel<<<cvtHH, blk256, 0, stream>>>(wo,  tw_wo, HID, HID);
    wt_convert_kernel<<<cvtHF, blk256, 0, stream>>>(w1,  tw_w1, HID, FFND);
    wt_convert_kernel<<<cvtFH, blk256, 0, stream>>>(w2,  tw_w2, FFND, HID);

    // q/k/v projections (q pre-scaled); v written transposed for PV WMMAs
    gemm_bf16_kernel<<<gemmGridH, blk256, 0, stream>>>(
        xb, HID, tw_q, HID, HID, bq, scale, 0, TOK, nullptr, qb, HID, 0);
    gemm_bf16_kernel<<<gemmGridH, blk256, 0, stream>>>(
        xb, HID, tw_k, HID, HID, bk, 1.0f, 0, TOK, nullptr, kb_, HID, 0);
    gemm_bf16_kernel<<<gemmGridH, blk256, 0, stream>>>(
        xb, HID, tw_v, HID, HID, bv, 1.0f, 0, TOK, nullptr, vTb, HID, 1);

    // global-attention projections
    gemm_bf16_kernel<<<gemmGridQG, blk256, 0, stream>>>(
        xb, (long)SEQ * HID, tw_qg, HID, HID, bqg, scale, 0, NBATCH,
        qgf, nullptr, HID, 0);
    gemm_bf16_kernel<<<gemmGridH, blk256, 0, stream>>>(
        xb, HID, tw_kg, HID, HID, bkg, 1.0f, 0, TOK, kgf, nullptr, HID, 0);
    gemm_bf16_kernel<<<gemmGridH, blk256, 0, stream>>>(
        xb, HID, tw_vg, HID, HID, bvg, 1.0f, 0, TOK, vgf, nullptr, HID, 0);

    // banded flash attention + global column
    flash_band_kernel<<<dim3(WSZ / 16, NBLK, NBATCH * NH), dim3(32), 0, stream>>>(
        qb, kb_, vTb, amask, attnf);

    // global attention for token 0
    global_attn_kernel<<<NBATCH * NH, blk256, 0, stream>>>(qgf, kgf, vgf, amask, ogf);

    // merge og into row 0 + bf16 mirror
    finalize_attn_kernel<<<(int)(((size_t)TOK * HID + 255) / 256), blk256, 0, stream>>>(
        attnf, attnb, ogf);

    // output projection (reuse kgf)
    gemm_bf16_kernel<<<gemmGridH, blk256, 0, stream>>>(
        attnb, HID, tw_wo, HID, HID, bo, 1.0f, 0, TOK, kgf, nullptr, HID, 0);

    // h = LN(x + attn)
    add_ln_kernel<<<TOK, blk256, 0, stream>>>(xf, kgf, l1g, l1b, hf, hb);

    // FFN: GELU(h @ w1 + b1) @ w2 + b2   (reuse vgf)
    gemm_bf16_kernel<<<gemmGridF, blk256, 0, stream>>>(
        hb, HID, tw_w1, HID, FFND, b1, 1.0f, 1, TOK, nullptr, ffnb, FFND, 0);
    gemm_bf16_kernel<<<gemmGridH, blk256, 0, stream>>>(
        ffnb, FFND, tw_w2, FFND, HID, b2, 1.0f, 0, TOK, vgf, nullptr, HID, 0);

    // x = LN(h + ffn)
    add_ln_kernel<<<TOK, blk256, 0, stream>>>(hf, vgf, l2g, l2b, xf, xb);
  }

  head_kernel<<<1, blk256, 0, stream>>>(xf, hw1, hb1, hw2, hb2, (float*)d_out);
}
